// Graph_encoder_29051158790635
// MI455X (gfx1250) — compile-verified, matrix operand order checked by reading, not run on hardware
//
#include <hip/hip_runtime.h>
#include <math.h>

#define NN 50000
#define EE 600000
#define BBATCH 32
#define CIN 128
#define CC 256
#define PADN (1 << 20)
#define BN_EPS 1e-5f
#define SLOPE 0.01f

typedef __attribute__((ext_vector_type(16))) __bf16          v16bf;
typedef __attribute__((ext_vector_type(16))) unsigned short  v16us;
typedef __attribute__((ext_vector_type(8)))  float           v8f;

__device__ __forceinline__ unsigned short f2bf(float f) {
  unsigned u = __float_as_uint(f);
  u += 0x7FFFu + ((u >> 16) & 1u);          // round-to-nearest-even
  return (unsigned short)(u >> 16);
}
__device__ __forceinline__ float lrelu(float x) { return x > 0.f ? x : SLOPE * x; }
// order-preserving float <-> uint for atomic max
__device__ __forceinline__ unsigned f2ord(float f) {
  unsigned u = __float_as_uint(f);
  return (u & 0x80000000u) ? ~u : (u | 0x80000000u);
}
__device__ __forceinline__ float ord2f(unsigned o) {
  unsigned u = (o & 0x80000000u) ? (o & 0x7FFFFFFFu) : ~o;
  return __uint_as_float(u);
}
// CDNA5 16-bit WMMA K-order within a 32-wide tile: [0..7,16..23 | 8..15,24..31]
// (maps aligned 4-element runs to aligned 4-element runs)
__device__ __forceinline__ int kperm(int c) {
  return (c < 8) ? c : (c < 16) ? (c + 8) : (c < 24) ? (c - 8) : c;
}
__device__ __forceinline__ v16bf ld_frag(const unsigned short* p) {
  v16us u = *reinterpret_cast<const v16us*>(p);   // 2x b128 load (LDS or global)
  union { v16us u; v16bf b; } c; c.u = u; return c.b;
}

// ---------------- fill / copy / convert ----------------
__global__ void k_fill_f(float* p, float v, int n) {
  int i = blockIdx.x * blockDim.x + threadIdx.x; if (i < n) p[i] = v;
}
__global__ void k_fill_i(int* p, int v, int n) {
  int i = blockIdx.x * blockDim.x + threadIdx.x; if (i < n) p[i] = v;
}
__global__ void k_fill_u32(unsigned* p, unsigned v, int n) {
  int i = blockIdx.x * blockDim.x + threadIdx.x; if (i < n) p[i] = v;
}
__global__ void k_fill_u64(unsigned long long* p, unsigned long long v, int n) {
  int i = blockIdx.x * blockDim.x + threadIdx.x; if (i < n) p[i] = v;
}
__global__ void k_copy_i(int* d, const int* s, int n) {
  int i = blockIdx.x * blockDim.x + threadIdx.x; if (i < n) d[i] = s[i];
}
// pack W[K,256] f32 row-major -> wpk bf16 in [col][K-tile x kperm] fragment order
__global__ void k_pack_w(unsigned short* wpk, const float* W, int K) {
  int i = blockIdx.x * blockDim.x + threadIdx.x;
  if (i >= K * CC) return;
  int r = i / CC, c = i % CC;
  wpk[(size_t)c * K + (r & ~31) + kperm(r & 31)] = f2bf(W[i]);
}

// ---------------- SAGE mean aggregation ----------------
__global__ void k_sage_agg(const float* __restrict__ x, const int* __restrict__ src,
                           const int* __restrict__ dst, const int* __restrict__ ev,
                           float* __restrict__ agg, float* __restrict__ deg, int C) {
  int e = blockIdx.x; int c = threadIdx.x;
  if (!ev[e]) return;
  int s = src[e], d = dst[e];
  atomicAdd(&agg[d * C + c], x[s * C + c]);
  if (c == 0) atomicAdd(&deg[d], 1.0f);
}
__global__ void k_agg_div(float* agg, const float* deg, int C, int n) {
  int i = blockIdx.x * blockDim.x + threadIdx.x; if (i >= n) return;
  agg[i] /= fmaxf(deg[i / C], 1.0f);
}

// ---------------- WMMA GEMM: out[M,256] = A[M,K] @ Wpk + (bias) (+beta*out)
// A f32 staged to LDS as bf16 in fragment order (wide path for full tiles);
// B (weights) pre-packed in global so each fragment is 2x global_load_b128.
__global__ __launch_bounds__(128)
void k_gemm_bias(const float* __restrict__ A, const unsigned short* __restrict__ Wpk,
                 const float* __restrict__ bias, float* __restrict__ out,
                 int M, int K, int beta) {
  __shared__ __align__(32) unsigned short lsA[64 * 32];  // [row][kperm]
  const int tid  = threadIdx.x;
  const int wave = tid >> 5;
  const int lane = tid & 31;
  const int half = lane >> 4;
  const int lm   = lane & 15;
  const int rowBase = blockIdx.x * 64;
  const int colBase = blockIdx.y * 64;
  const bool fullTile = (rowBase + 64 <= M);

  v8f acc[4];
  for (int t = 0; t < 4; ++t)
    for (int j = 0; j < 8; ++j) acc[t][j] = 0.f;

  for (int kk = 0; kk < K; kk += 32) {
    // stage A tile 64x32 (f32 -> bf16), K pre-swizzled
    if (fullTile) {
      for (int i = tid; i < 64 * 8; i += 128) {          // 512 float4s
        int r = i >> 3, c4 = (i & 7) * 4;
        const float4 v = *(const float4*)&A[(size_t)(rowBase + r) * K + kk + c4];
        unsigned p0 = (unsigned)f2bf(v.x) | ((unsigned)f2bf(v.y) << 16);
        unsigned p1 = (unsigned)f2bf(v.z) | ((unsigned)f2bf(v.w) << 16);
        unsigned long long pk = (unsigned long long)p0 | ((unsigned long long)p1 << 32);
        *(unsigned long long*)&lsA[r * 32 + kperm(c4)] = pk;
      }
    } else {
      for (int i = tid; i < 64 * 32; i += 128) {
        int r = i >> 5, c = i & 31;
        int gr = rowBase + r;
        float v = (gr < M) ? A[(size_t)gr * K + kk + c] : 0.f;
        lsA[r * 32 + kperm(c)] = f2bf(v);
      }
    }
    if (kk + 32 < K && fullTile)
      __builtin_prefetch(&A[(size_t)(rowBase + (tid >> 1)) * K + kk + 32], 0, 1);
    __syncthreads();

    v16bf afrag = ld_frag(&lsA[(wave * 16 + lm) * 32 + half * 16]);
#pragma unroll
    for (int nt = 0; nt < 4; ++nt) {
      int col = colBase + nt * 16 + lm;
      v16bf bfrag = ld_frag(&Wpk[(size_t)col * K + kk + half * 16]);
      acc[nt] = __builtin_amdgcn_wmma_f32_16x16x32_bf16(
          false, afrag, false, bfrag, (short)0, acc[nt], false, false);
    }
    __syncthreads();
  }

  // epilogue: C/D layout — lanes 0-15: M=r, lanes 16-31: M=r+8; N=lane%16
#pragma unroll
  for (int nt = 0; nt < 4; ++nt) {
    int col = colBase + nt * 16 + lm;
    float bs = bias ? bias[col] : 0.f;
#pragma unroll
    for (int r = 0; r < 8; ++r) {
      int row = rowBase + wave * 16 + half * 8 + r;
      if (row < M) {
        float v = acc[nt][r] + bs;
        if (beta) v += out[(size_t)row * CC + col];
        out[(size_t)row * CC + col] = v;
      }
    }
  }
}

// ---------------- L2 normalize per node, mask by nv ----------------
__global__ __launch_bounds__(256)
void k_l2norm(float* x, const int* nv) {
  __shared__ float sd[256];
  int i = blockIdx.x; int c = threadIdx.x;
  float v = x[(size_t)i * CC + c];
  sd[c] = v * v;
  __syncthreads();
  for (int s = 128; s > 0; s >>= 1) {
    if (c < s) sd[c] += sd[c + s];
    __syncthreads();
  }
  float scale = nv[i] ? (1.f / fmaxf(sqrtf(sd[0]), 1e-12f)) : 0.f;
  x[(size_t)i * CC + c] = v * scale;
}

// ---------------- masked batch norm ----------------
__global__ void k_bn_stats(const float* x, const int* nv, float* stats, int n) {
  int i = blockIdx.x * blockDim.x + threadIdx.x; if (i >= n) return;
  int node = i / CC, c = i % CC;
  if (!nv[node]) return;
  float v = x[i];
  atomicAdd(&stats[c], v);
  atomicAdd(&stats[CC + c], v * v);
  if (c == 0) atomicAdd(&stats[2 * CC], 1.0f);
}
__global__ void k_bn_apply(float* x, const int* nv, const float* stats,
                           const float* g, const float* b, int n) {
  int i = blockIdx.x * blockDim.x + threadIdx.x; if (i >= n) return;
  int node = i / CC, c = i % CC;
  float cnt  = fmaxf(stats[2 * CC], 1.0f);
  float mean = stats[c] / cnt;
  float var  = stats[CC + c] / cnt - mean * mean;
  float v = (x[i] - mean) * rsqrtf(fmaxf(var, 0.f) + BN_EPS) * g[c] + b[c];
  x[i] = nv[node] ? lrelu(v) : 0.f;
}

// ---------------- EdgePooling ----------------
__global__ __launch_bounds__(256)
void k_edge_raw(const float* __restrict__ h, const int* __restrict__ src,
                const int* __restrict__ dst, const float* __restrict__ We,
                const float* __restrict__ be, float* __restrict__ raw) {
  int e = blockIdx.x * 8 + (threadIdx.x >> 5);
  int lane = threadIdx.x & 31;
  if (e >= EE) return;
  int s = src[e], d = dst[e];
  float acc = 0.f;
  for (int c = lane; c < CC; c += 32)
    acc += h[(size_t)s * CC + c] * We[c] + h[(size_t)d * CC + c] * We[CC + c];
  for (int off = 16; off > 0; off >>= 1) acc += __shfl_xor(acc, off, 32);
  if (lane == 0) raw[e] = acc + be[0];
}
__global__ void k_seg_max(const float* raw, const int* dst, const int* ev, unsigned* mxb) {
  int e = blockIdx.x * blockDim.x + threadIdx.x; if (e >= EE) return;
  if (!ev[e]) return;
  atomicMax(&mxb[dst[e]], f2ord(raw[e]));
}
__global__ void k_edge_exp(const float* raw, const int* dst, const int* ev,
                           const unsigned* mxb, float* den, float* evl) {
  int e = blockIdx.x * blockDim.x + threadIdx.x; if (e >= EE) return;
  float w = 0.f;
  if (ev[e]) {
    float mx = ord2f(mxb[dst[e]]);
    if (mx < -1e37f) mx = 0.f;
    w = __expf(raw[e] - mx);
    atomicAdd(&den[dst[e]], w);
  }
  evl[e] = w;
}
__global__ void k_edge_score(const int* dst, const int* ev, const float* den,
                             float* escore, unsigned long long* keys) {
  int e = blockIdx.x * blockDim.x + threadIdx.x; if (e >= EE) return;
  float sc = escore[e] / fmaxf(den[dst[e]], 1e-16f) + 0.5f;
  escore[e] = sc;
  float kf = ev[e] ? sc : -INFINITY;
  keys[e] = ((unsigned long long)(~f2ord(kf)) << 32) | (unsigned)e;  // descending score
}
__global__ void k_bitonic(unsigned long long* key, int j, int k) {
  int i = blockIdx.x * blockDim.x + threadIdx.x;
  int ixj = i ^ j;
  if (ixj <= i) return;
  unsigned long long a = key[i], b = key[ixj];
  bool up = ((i & k) == 0);
  if ((a > b) == up) { key[i] = b; key[ixj] = a; }
}
__global__ void k_greedy(const unsigned long long* keys, const int* src, const int* dst,
                         const int* ev, int* used, int* chosen) {
  if (threadIdx.x != 0 || blockIdx.x != 0) return;
  for (int t = 0; t < EE; ++t) {
    unsigned idx = (unsigned)(keys[t] & 0xFFFFFFFFu);
    int i = src[idx], j = dst[idx];
    int can = ev[idx] && !used[i] && !used[j] && (i != j);
    if (can) { used[i] = 1; used[j] = 1; }
    chosen[idx] = can;
  }
}
__global__ void k_cluster_init(int* cluster, float* rep) {
  int i = blockIdx.x * blockDim.x + threadIdx.x;
  if (i < NN) { cluster[i] = i; rep[i] = 1.f; }
}
__global__ void k_apply_chosen(const int* chosen, const int* src, const int* dst,
                               const float* escore, int* cluster, float* rep) {
  int e = blockIdx.x * blockDim.x + threadIdx.x; if (e >= EE) return;
  if (chosen[e]) { cluster[dst[e]] = src[e]; rep[src[e]] = escore[e]; }
}
__global__ void k_counts(const int* cluster, const int* nv, float* counts) {
  int i = blockIdx.x * blockDim.x + threadIdx.x; if (i >= NN) return;
  if (nv[i]) atomicAdd(&counts[cluster[i]], 1.f);
}
__global__ void k_pool_sum(const float* h, const int* cluster, const int* nv, float* nf) {
  int i = blockIdx.x; int c = threadIdx.x;
  if (!nv[i]) return;
  atomicAdd(&nf[(size_t)cluster[i] * CC + c], h[(size_t)i * CC + c]);
}
__global__ void k_pool_scale(float* nf, const float* rep, const float* counts, int* nv) {
  int i = blockIdx.x; int c = threadIdx.x;
  int valid = counts[i] > 0.f;
  nf[(size_t)i * CC + c] *= rep[i] * (valid ? 1.f : 0.f);
  if (c == 0) nv[i] = valid;
}
__global__ void k_remap_keys(const int* src, const int* dst, const int* ev,
                             const int* cluster, unsigned long long* keys) {
  int e = blockIdx.x * blockDim.x + threadIdx.x; if (e >= EE) return;
  unsigned long long s2  = (unsigned)cluster[src[e]];
  unsigned long long d2  = (unsigned)cluster[dst[e]];
  unsigned long long inv = ev[e] ? 0ULL : 1ULL;
  keys[e] = (inv << 52) | (s2 << 36) | (d2 << 20) | (unsigned long long)e;
}
__global__ void k_dedup(const unsigned long long* keys, int* src, int* dst, int* ev) {
  int t = blockIdx.x * blockDim.x + threadIdx.x; if (t >= EE) return;
  unsigned long long k = keys[t];
  int s2 = (int)((k >> 36) & 0xFFFF), d2 = (int)((k >> 20) & 0xFFFF);
  int vv = !((k >> 52) & 1);
  int dup = 0;
  if (t > 0) {
    unsigned long long kp = keys[t - 1];
    int sp = (int)((kp >> 36) & 0xFFFF), dp = (int)((kp >> 20) & 0xFFFF);
    int vp = !((kp >> 52) & 1);
    dup = (sp == s2) && (dp == d2) && vv && vp;
  }
  src[t] = s2; dst[t] = d2; ev[t] = vv && !dup;
}

// ---------------- readout ----------------
__global__ void k_batch_pool(const float* h, const int* batch, float* pooled) {
  int i = blockIdx.x, c = threadIdx.x;
  atomicAdd(&pooled[(size_t)batch[i] * CC + c], h[(size_t)i * CC + c]);
}
__global__ void k_final_bn(const float* x, const float* g, const float* b, float* out) {
  int c = threadIdx.x;
  float s = 0.f, sq = 0.f;
  for (int r = 0; r < BBATCH; ++r) { float v = x[r * CC + c]; s += v; sq += v * v; }
  float mean = s / BBATCH;
  float var  = sq / BBATCH - mean * mean;
  float inv  = rsqrtf(fmaxf(var, 0.f) + BN_EPS);
  for (int r = 0; r < BBATCH; ++r) {
    float v = (x[r * CC + c] - mean) * inv * g[c] + b[c];
    out[r * CC + c] = lrelu(v);
  }
}

extern "C" void kernel_launch(void* const* d_in, const int* in_sizes, int n_in,
                              void* d_out, int out_size, void* d_ws, size_t ws_size,
                              hipStream_t stream) {
  (void)in_sizes; (void)n_in; (void)out_size; (void)ws_size;
  const float* x     = (const float*)d_in[0];
  const int*   src0  = (const int*)d_in[1];
  const int*   dst0  = (const int*)d_in[2];
  const int*   batch = (const int*)d_in[3];
  auto P = [&](int i) { return (const float*)d_in[i]; };

  char* base = (char*)d_ws; size_t off = 0;
  auto alloc = [&](size_t bytes) -> void* {
    off = (off + 255) & ~(size_t)255;
    void* p = base + off; off += bytes; return p;
  };
  float* buf0 = (float*)alloc(sizeof(float) * (size_t)NN * CC);
  float* buf1 = (float*)alloc(sizeof(float) * (size_t)NN * CC);
  float* aggf = (float*)alloc(sizeof(float) * (size_t)NN * CC);
  unsigned short* wbfL = (unsigned short*)alloc(sizeof(unsigned short) * CC * CC);
  unsigned short* wbfR = (unsigned short*)alloc(sizeof(unsigned short) * CC * CC);
  float* stats  = (float*)alloc(sizeof(float) * (2 * CC + 1));
  float* deg    = (float*)alloc(sizeof(float) * NN);
  int*   nv     = (int*)alloc(sizeof(int) * NN);
  int*   esrc   = (int*)alloc(sizeof(int) * EE);
  int*   edst   = (int*)alloc(sizeof(int) * EE);
  int*   eev    = (int*)alloc(sizeof(int) * EE);
  float* raw    = (float*)alloc(sizeof(float) * EE);
  float* escore = (float*)alloc(sizeof(float) * EE);
  unsigned* mxb = (unsigned*)alloc(sizeof(unsigned) * NN);
  float* den    = (float*)alloc(sizeof(float) * NN);
  unsigned long long* keys = (unsigned long long*)alloc(sizeof(unsigned long long) * PADN);
  int*   used   = (int*)alloc(sizeof(int) * NN);
  int*   chosen = (int*)alloc(sizeof(int) * EE);
  int*   cluster= (int*)alloc(sizeof(int) * NN);
  float* rep    = (float*)alloc(sizeof(float) * NN);
  float* counts = (float*)alloc(sizeof(float) * NN);
  float* pooled = (float*)alloc(sizeof(float) * BBATCH * CC);
  float* outt   = (float*)alloc(sizeof(float) * BBATCH * CC);

  const int TPB = 256;
  auto gs = [&](int n) { return (n + TPB - 1) / TPB; };

  k_copy_i<<<gs(EE), TPB, 0, stream>>>(esrc, src0, EE);
  k_copy_i<<<gs(EE), TPB, 0, stream>>>(edst, dst0, EE);
  k_fill_i<<<gs(EE), TPB, 0, stream>>>(eev, 1, EE);
  k_fill_i<<<gs(NN), TPB, 0, stream>>>(nv, 1, NN);

  auto sage = [&](const float* fin, float* fout, int K,
                  const float* Wl, const float* bl, const float* Wr,
                  const float* g, const float* b) {
    k_pack_w<<<gs(K * CC), TPB, 0, stream>>>(wbfL, Wl, K);
    k_pack_w<<<gs(K * CC), TPB, 0, stream>>>(wbfR, Wr, K);
    k_fill_f<<<gs(NN * K), TPB, 0, stream>>>(aggf, 0.f, NN * K);
    k_fill_f<<<gs(NN), TPB, 0, stream>>>(deg, 0.f, NN);
    k_sage_agg<<<EE, K, 0, stream>>>(fin, esrc, edst, eev, aggf, deg, K);
    k_agg_div<<<gs(NN * K), TPB, 0, stream>>>(aggf, deg, K, NN * K);
    dim3 gg((NN + 63) / 64, 4);
    k_gemm_bias<<<gg, 128, 0, stream>>>(aggf, wbfL, bl, fout, NN, K, 0);
    k_gemm_bias<<<gg, 128, 0, stream>>>(fin, wbfR, nullptr, fout, NN, K, 1);
    k_l2norm<<<NN, CC, 0, stream>>>(fout, nv);
    k_fill_f<<<gs(2 * CC + 1), TPB, 0, stream>>>(stats, 0.f, 2 * CC + 1);
    k_bn_stats<<<gs(NN * CC), TPB, 0, stream>>>(fout, nv, stats, NN * CC);
    k_bn_apply<<<gs(NN * CC), TPB, 0, stream>>>(fout, nv, stats, g, b, NN * CC);
  };

  auto sortkeys = [&]() {
    for (int k = 2; k <= PADN; k <<= 1)
      for (int j = k >> 1; j > 0; j >>= 1)
        k_bitonic<<<PADN / TPB, TPB, 0, stream>>>(keys, j, k);
  };

  auto edgepool = [&](const float* hin, float* hout, const float* We, const float* be) {
    k_edge_raw<<<(EE + 7) / 8, 256, 0, stream>>>(hin, esrc, edst, We, be, raw);
    k_fill_u32<<<gs(NN), TPB, 0, stream>>>(mxb, 0x007FFFFFu, NN);  // ord(-inf)
    k_seg_max<<<gs(EE), TPB, 0, stream>>>(raw, edst, eev, mxb);
    k_fill_f<<<gs(NN), TPB, 0, stream>>>(den, 0.f, NN);
    k_edge_exp<<<gs(EE), TPB, 0, stream>>>(raw, edst, eev, mxb, den, escore);
    k_fill_u64<<<gs(PADN), TPB, 0, stream>>>(keys, ~0ULL, PADN);
    k_edge_score<<<gs(EE), TPB, 0, stream>>>(edst, eev, den, escore, keys);
    sortkeys();
    k_fill_i<<<gs(NN), TPB, 0, stream>>>(used, 0, NN);
    k_greedy<<<1, 1, 0, stream>>>(keys, esrc, edst, eev, used, chosen);
    k_cluster_init<<<gs(NN), TPB, 0, stream>>>(cluster, rep);
    k_apply_chosen<<<gs(EE), TPB, 0, stream>>>(chosen, esrc, edst, escore, cluster, rep);
    k_fill_f<<<gs(NN), TPB, 0, stream>>>(counts, 0.f, NN);
    k_counts<<<gs(NN), TPB, 0, stream>>>(cluster, nv, counts);
    k_fill_f<<<gs(NN * CC), TPB, 0, stream>>>(hout, 0.f, NN * CC);
    k_pool_sum<<<NN, CC, 0, stream>>>(hin, cluster, nv, hout);
    k_pool_scale<<<NN, CC, 0, stream>>>(hout, rep, counts, nv);
    k_fill_u64<<<gs(PADN), TPB, 0, stream>>>(keys, ~0ULL, PADN);
    k_remap_keys<<<gs(EE), TPB, 0, stream>>>(esrc, edst, eev, cluster, keys);
    sortkeys();
    k_dedup<<<gs(EE), TPB, 0, stream>>>(keys, esrc, edst, eev);
  };

  // layer params: l=1..4 at indices 4+(l-1)*5; We/be at 24+..; head at 30..33
  sage(x,    buf0, CIN, P(4),  P(5),  P(6),  P(7),  P(8));
  edgepool(buf0, buf1, P(24), P(25));
  sage(buf1, buf0, CC,  P(9),  P(10), P(11), P(12), P(13));
  edgepool(buf0, buf1, P(26), P(27));
  sage(buf1, buf0, CC,  P(14), P(15), P(16), P(17), P(18));
  edgepool(buf0, buf1, P(28), P(29));
  sage(buf1, buf0, CC,  P(19), P(20), P(21), P(22), P(23));

  k_fill_f<<<gs(BBATCH * CC), TPB, 0, stream>>>(pooled, 0.f, BBATCH * CC);
  k_batch_pool<<<NN, CC, 0, stream>>>(buf0, batch, pooled);
  k_pack_w<<<gs(CC * CC), TPB, 0, stream>>>(wbfL, P(30), CC);
  dim3 gf(1, 4);
  k_gemm_bias<<<gf, 128, 0, stream>>>(pooled, wbfL, P(31), outt, BBATCH, CC, 0);
  k_final_bn<<<1, CC, 0, stream>>>(outt, P(32), P(33), (float*)d_out);
}
